// Model_40836549050648
// MI455X (gfx1250) — matrix operand order repaired and checked by reading, not compiled, not run on hardware
//
#include <hip/hip_runtime.h>

// ---------------------------------------------------------------------------
// Model: RevIN -> channel MLP -> cosine routing + threefry bernoulli mask ->
// masked exp-attention over K=16 clusters -> 4x mixer (LN+GEMM+GELU+res) ->
// gated per-channel output projection -> RevIN denorm.
// B=32, L=C=H=512, K=16, P=96, NB=4.
//
// All 16 big GEMMs are [16384,512]x[512,512]:
//   v_wmma_f32_16x16x32_bf16 (bf16 in, f32 accumulate)
//   global_load_async_to_lds_b128 double-buffered tile staging (ASYNCcnt)
// Weights are pre-transposed to [N,K] at bf16-conversion time so both
// operands stream K-contiguously into LDS with no in-loop transpose.
// ---------------------------------------------------------------------------

typedef __attribute__((ext_vector_type(16))) __bf16 v16bf;
typedef __attribute__((ext_vector_type(8)))  float  v8f;

union FragBF { v16bf v; unsigned u[8]; };

__device__ __forceinline__ unsigned short f2bf(float f) {
  unsigned u = __float_as_uint(f);
  unsigned rnd = 0x7FFFu + ((u >> 16) & 1u);   // round-to-nearest-even
  return (unsigned short)((u + rnd) >> 16);
}

__device__ __forceinline__ float blockReduceSum256(float v, float* red) {
  int t = threadIdx.x;
  red[t] = v; __syncthreads();
  for (int s = 128; s > 0; s >>= 1) { if (t < s) red[t] += red[t + s]; __syncthreads(); }
  float r = red[0]; __syncthreads();
  return r;
}

__device__ __forceinline__ void blockReduce2_256(float a, float b, float* r1, float* r2,
                                                 float& oa, float& ob) {
  int t = threadIdx.x;
  r1[t] = a; r2[t] = b; __syncthreads();
  for (int s = 128; s > 0; s >>= 1) {
    if (t < s) { r1[t] += r1[t + s]; r2[t] += r2[t + s]; }
    __syncthreads();
  }
  oa = r1[0]; ob = r2[0]; __syncthreads();
}

// ---------------------------------------------------------------------------
// f32 [K,N] -> bf16 [N,K] transpose-convert (weights; tiny, done once)
// ---------------------------------------------------------------------------
__global__ void convT_bf16_kern(const float* __restrict__ in,
                                unsigned short* __restrict__ out) {
  int idx = blockIdx.x * 256 + threadIdx.x;   // over 512*512
  int k = idx >> 9, n = idx & 511;
  out[(size_t)n * 512 + k] = f2bf(in[idx]);
}

// ---------------------------------------------------------------------------
// RevIN: stats over time dim + normalize + transpose to [B,C,L] bf16
// ---------------------------------------------------------------------------
__global__ __launch_bounds__(256)
void revin_kern(const float* __restrict__ x, const float* __restrict__ rev_w,
                const float* __restrict__ rev_b, float* __restrict__ meanO,
                float* __restrict__ stdO, unsigned short* __restrict__ xt_bf) {
  __shared__ float r1[256], r2[256];
  int bc = blockIdx.x;                 // b*512 + c
  int b = bc >> 9, c = bc & 511;
  int t = threadIdx.x;
  float v0 = x[((size_t)b * 512 + t) * 512 + c];
  float v1 = x[((size_t)b * 512 + t + 256) * 512 + c];
  float S1, S2;
  blockReduce2_256(v0 + v1, v0 * v0 + v1 * v1, r1, r2, S1, S2);
  float mean = S1 * (1.0f / 512.0f);
  float var = S2 * (1.0f / 512.0f) - mean * mean;
  float sd = sqrtf(var + 1e-5f);
  float istd = 1.0f / sd;
  if (t == 0) { meanO[bc] = mean; stdO[bc] = sd; }
  float g = rev_w[c], bb = rev_b[c];
  xt_bf[(size_t)bc * 512 + t]       = f2bf((v0 - mean) * istd * g + bb);
  xt_bf[(size_t)bc * 512 + t + 256] = f2bf((v1 - mean) * istd * g + bb);
}

// ---------------------------------------------------------------------------
// Row LayerNorm over last dim (512) -> bf16 output
// ---------------------------------------------------------------------------
__global__ __launch_bounds__(256)
void ln_rows_kern(const float* __restrict__ in, const float* __restrict__ g,
                  const float* __restrict__ b2, unsigned short* __restrict__ outH) {
  __shared__ float r1[256], r2[256];
  int row = blockIdx.x, t = threadIdx.x;
  const float* rp = in + (size_t)row * 512;
  float v0 = rp[t], v1 = rp[t + 256];
  float S1, S2;
  blockReduce2_256(v0 + v1, v0 * v0 + v1 * v1, r1, r2, S1, S2);
  float mean = S1 * (1.0f / 512.0f);
  float var = S2 * (1.0f / 512.0f) - mean * mean;
  float istd = rsqrtf(var + 1e-5f);
  outH[(size_t)row * 512 + t]       = f2bf((v0 - mean) * istd * g[t] + b2[t]);
  outH[(size_t)row * 512 + t + 256] = f2bf((v1 - mean) * istd * g[t + 256] + b2[t + 256]);
}

// ---------------------------------------------------------------------------
// Per-batch 512x512 transpose: out[b][y][x] = in[b][x][y]
// ---------------------------------------------------------------------------
__global__ void transpose512_kern(const float* __restrict__ in, float* __restrict__ out) {
  __shared__ float tile[32][33];
  int b = blockIdx.z;
  int x0 = blockIdx.x * 32, y0 = blockIdx.y * 32;
  const float* src = in + (size_t)b * 512 * 512;
  float* dst = out + (size_t)b * 512 * 512;
  int tx = threadIdx.x, ty = threadIdx.y;
  for (int j = 0; j < 32; j += 8)
    tile[ty + j][tx] = src[(size_t)(y0 + ty + j) * 512 + x0 + tx];
  __syncthreads();
  for (int j = 0; j < 32; j += 8)
    dst[(size_t)(x0 + ty + j) * 512 + y0 + tx] = tile[tx][ty + j];
}

// ---------------------------------------------------------------------------
// Async Global->LDS tile copy: 128 rows x 32 k (bf16), source row-major with
// row stride 512 elements; LDS row stride 112 bytes (16B aligned; 28 dwords
// per row => conflict-free fragment reads). Each of 8 waves issues 2
// global_load_async_to_lds_b128 ops (16B/lane). Tracked by ASYNCcnt.
// ---------------------------------------------------------------------------
__device__ __forceinline__ void async_tile_load(const unsigned short* src, int row0, int k0,
                                                unsigned ldsBase, int wave, int lane) {
  int rsub = lane >> 2;       // 0..7  (row within 8-row chunk)
  int kq = lane & 3;          // 16B quarter of the 64B k-row
#pragma unroll
  for (int c = 0; c < 2; ++c) {
    int row = (wave * 2 + c) * 8 + rsub;      // 0..127
    const void* g = (const char*)src + ((size_t)(row0 + row) * 512 + k0) * 2 + kq * 16;
    unsigned l = ldsBase + (unsigned)(row * 112 + kq * 16);
    asm volatile("global_load_async_to_lds_b128 %0, %1, off"
                 :: "v"(l), "v"(g) : "memory");
  }
}

__device__ __forceinline__ void wait_async0() {
  asm volatile("s_wait_asynccnt 0" ::: "memory");
}

// ---------------------------------------------------------------------------
// bf16 WMMA GEMM: D[16384,512] = act(A[16384,512] * W^T + bias) (+ res)
// A row-major [M,K] bf16; Bt pre-transposed weights [N,K] bf16.
// Block tile 128x128, 8 wave32 waves, wave tile 32x64 (2x4 WMMA frags),
// K-step 32, double-buffered async LDS staging.
// ACT: 0 none, 1 relu, 2 exact gelu.
// ---------------------------------------------------------------------------
template <int ACT, bool RES, bool OUTF, bool OUTH>
__global__ __launch_bounds__(256)
void gemm_kern(const unsigned short* __restrict__ A,
               const unsigned short* __restrict__ Bt,
               const float* __restrict__ bias,
               const float* __restrict__ res,
               float* __restrict__ outF,
               unsigned short* __restrict__ outH) {
  __shared__ __align__(16) unsigned short As[2][128 * 56];
  __shared__ __align__(16) unsigned short Bs[2][128 * 56];
  const int N = 512;
  int tid = threadIdx.x;
  int wave = tid >> 5, lane = tid & 31;
  int half = lane >> 4, r16 = lane & 15;
  int wm = (wave >> 1) << 5;          // 0,32,64,96
  int wn = (wave & 1) << 6;           // 0,64
  int m0 = blockIdx.y << 7, n0 = blockIdx.x << 7;

  unsigned asBase[2] = { (unsigned)(unsigned long long)(void*)&As[0][0],
                         (unsigned)(unsigned long long)(void*)&As[1][0] };
  unsigned bsBase[2] = { (unsigned)(unsigned long long)(void*)&Bs[0][0],
                         (unsigned)(unsigned long long)(void*)&Bs[1][0] };

  v8f acc[2][4] = {};

  __builtin_prefetch(A + (size_t)m0 * 512, 0, 1);

  // prologue: stage k=0 tiles
  async_tile_load(A, m0, 0, asBase[0], wave, lane);
  async_tile_load(Bt, n0, 0, bsBase[0], wave, lane);
  wait_async0();
  __syncthreads();

  for (int k0 = 0; k0 < 512; k0 += 32) {
    int cur = (k0 >> 5) & 1, nxt = cur ^ 1;
    if (k0 + 32 < 512) {                       // overlap next tile with compute
      async_tile_load(A, m0, k0 + 32, asBase[nxt], wave, lane);
      async_tile_load(Bt, n0, k0 + 32, bsBase[nxt], wave, lane);
    }

    FragBF afr[2], bfr[4];
#pragma unroll
    for (int mi = 0; mi < 2; ++mi) {
      // A 16x32 bf16 layout: lane half 0 -> K {0..7,16..23}, half 1 -> {8..15,24..31}
      const unsigned* src = (const unsigned*)(As[cur] + (wm + (mi << 4) + r16) * 56);
#pragma unroll
      for (int i = 0; i < 4; ++i) {
        afr[mi].u[i]     = src[i + (half << 2)];
        afr[mi].u[4 + i] = src[8 + i + (half << 2)];
      }
    }
#pragma unroll
    for (int ni = 0; ni < 4; ++ni) {
      // B 32x16 bf16 layout: lane half 0 -> K {0..7,16..23}, half 1 -> {8..15,24..31}
      const unsigned* src = (const unsigned*)(Bs[cur] + (wn + (ni << 4) + r16) * 56);
#pragma unroll
      for (int i = 0; i < 4; ++i) {
        bfr[ni].u[i]     = src[i + (half << 2)];
        bfr[ni].u[4 + i] = src[8 + i + (half << 2)];
      }
    }
#pragma unroll
    for (int mi = 0; mi < 2; ++mi)
#pragma unroll
      for (int ni = 0; ni < 4; ++ni)
        acc[mi][ni] = __builtin_amdgcn_wmma_f32_16x16x32_bf16(
            false, afr[mi].v, false, bfr[ni].v, (short)0, acc[mi][ni], false, false);

    wait_async0();
    __syncthreads();
  }

  // epilogue: C/D layout -> vgpr d, lanes 0-15: M=d, lanes 16-31: M=8+d; N=r16
#pragma unroll
  for (int mi = 0; mi < 2; ++mi) {
#pragma unroll
    for (int ni = 0; ni < 4; ++ni) {
      int nn = n0 + wn + (ni << 4) + r16;
      float bb = bias[nn];
#pragma unroll
      for (int d = 0; d < 8; ++d) {
        int mm = m0 + wm + (mi << 4) + (half << 3) + d;
        float v = acc[mi][ni][d] + bb;
        if (ACT == 1) v = fmaxf(v, 0.0f);
        if (ACT == 2) v = 0.5f * v * (1.0f + erff(v * 0.7071067811865475f));
        size_t o = (size_t)mm * N + nn;
        if (RES)  v += res[o];
        if (OUTF) outF[o] = v;
        if (OUTH) outH[o] = f2bf(v);
      }
    }
  }
}

// ---------------------------------------------------------------------------
// Normalize cluster embeddings: cn[k] = ce[k] / max(||ce[k]||, 1e-12)
// ---------------------------------------------------------------------------
__global__ __launch_bounds__(256)
void cn_kern(const float* __restrict__ ce, float* __restrict__ cn) {
  __shared__ float red[256];
  int k = blockIdx.x, t = threadIdx.x;
  float v0 = ce[k * 512 + t], v1 = ce[k * 512 + t + 256];
  float ss = blockReduceSum256(v0 * v0 + v1 * v1, red);
  float inv = 1.0f / fmaxf(sqrtf(ss), 1e-12f);
  cn[k * 512 + t] = v0 * inv;
  cn[k * 512 + t + 256] = v1 * inv;
}

// ---------------------------------------------------------------------------
// Routing: p[b,c,:] = softmax over K of (h[b,c]/||h||) . cn[k]
// ---------------------------------------------------------------------------
__global__ __launch_bounds__(256)
void route_kern(const float* __restrict__ h, const float* __restrict__ cn,
                float* __restrict__ p) {
  __shared__ float hr[512];
  __shared__ float red[256];
  __shared__ float sc16[16];
  int bc = blockIdx.x, t = threadIdx.x;
  const float* row = h + (size_t)bc * 512;
  float v0 = row[t], v1 = row[t + 256];
  hr[t] = v0; hr[t + 256] = v1;
  float ss = blockReduceSum256(v0 * v0 + v1 * v1, red);
  float inv = 1.0f / fmaxf(sqrtf(ss), 1e-12f);
  if (t < 16) {
    float d = 0.0f;
    const float* cr = cn + t * 512;
    for (int hh = 0; hh < 512; ++hh) d += hr[hh] * cr[hh];
    sc16[t] = d * inv;
  }
  __syncthreads();
  if (t < 16) {
    float mx = -1e30f;
    for (int i = 0; i < 16; ++i) mx = fmaxf(mx, sc16[i]);
    float s = 0.0f;
    for (int i = 0; i < 16; ++i) s += expf(sc16[i] - mx);
    p[(size_t)bc * 16 + t] = expf(sc16[t] - mx) / s;
  }
}

// ---------------------------------------------------------------------------
// Bernoulli mask, exact JAX threefry2x32 with key(42) = {0, 42}.
// ---------------------------------------------------------------------------
#define TFR(x0, x1, r) { x0 += x1; x1 = (x1 << r) | (x1 >> (32 - r)); x1 ^= x0; }
__global__ void bernoulli_kern(const float* __restrict__ p, float* __restrict__ M, int half) {
  int i = blockIdx.x * blockDim.x + threadIdx.x;
  if (i >= half) return;
  unsigned x0 = (unsigned)i, x1 = (unsigned)(i + half);
  const unsigned ks0 = 0u, ks1 = 42u, ks2 = 0x1BD11BDAu ^ 0u ^ 42u;
  x0 += ks0; x1 += ks1;
  TFR(x0, x1, 13) TFR(x0, x1, 15) TFR(x0, x1, 26) TFR(x0, x1, 6)
  x0 += ks1; x1 += ks2 + 1u;
  TFR(x0, x1, 17) TFR(x0, x1, 29) TFR(x0, x1, 16) TFR(x0, x1, 24)
  x0 += ks2; x1 += ks0 + 2u;
  TFR(x0, x1, 13) TFR(x0, x1, 15) TFR(x0, x1, 26) TFR(x0, x1, 6)
  x0 += ks0; x1 += ks1 + 3u;
  TFR(x0, x1, 17) TFR(x0, x1, 29) TFR(x0, x1, 16) TFR(x0, x1, 24)
  x0 += ks1; x1 += ks2 + 4u;
  TFR(x0, x1, 13) TFR(x0, x1, 15) TFR(x0, x1, 26) TFR(x0, x1, 6)
  x0 += ks2; x1 += ks0 + 5u;
  float u0 = __uint_as_float((x0 >> 9) | 0x3f800000u) - 1.0f;
  float u1 = __uint_as_float((x1 >> 9) | 0x3f800000u) - 1.0f;
  M[i]        = (u0 < p[i])        ? 1.0f : 0.0f;
  M[i + half] = (u1 < p[i + half]) ? 1.0f : 0.0f;
}

// ---------------------------------------------------------------------------
// Q = cluster_embeds @ wq + bq   [16,512]
// ---------------------------------------------------------------------------
__global__ __launch_bounds__(256)
void qproj_kern(const float* __restrict__ ce, const float* __restrict__ wq,
                const float* __restrict__ bq, float* __restrict__ Q) {
  __shared__ float cr[512];
  int k = blockIdx.x, t = threadIdx.x;
  cr[t] = ce[k * 512 + t]; cr[t + 256] = ce[k * 512 + t + 256];
  __syncthreads();
  for (int j = t; j < 512; j += 256) {
    float a = bq[j];
    for (int hh = 0; hh < 512; ++hh) a += cr[hh] * wq[(size_t)hh * 512 + j];
    Q[k * 512 + j] = a;
  }
}

// scores[b,k,c] = Q[k].Kk[b,c] / sqrt(512)
__global__ __launch_bounds__(256)
void scores_kern(const float* __restrict__ Q, const float* __restrict__ Kk,
                 float* __restrict__ sc) {
  __shared__ float qs[512];
  int bk = blockIdx.x;
  int b = bk >> 4, k = bk & 15;
  int t = threadIdx.x;
  qs[t] = Q[k * 512 + t]; qs[t + 256] = Q[k * 512 + t + 256];
  __syncthreads();
  for (int c = t; c < 512; c += 256) {
    const float* kr = Kk + ((size_t)b * 512 + c) * 512;
    float s = 0.0f;
    for (int hh = 0; hh < 512; ++hh) s += qs[hh] * kr[hh];
    sc[(size_t)bk * 512 + c] = s * 0.04419417382415922f;  // 1/sqrt(512)
  }
}

// aw = exp(scores)*M^T normalized over c (in-place in sc)
__global__ __launch_bounds__(256)
void aw_kern(float* __restrict__ sc, const float* __restrict__ Mm) {
  __shared__ float red[256];
  int bk = blockIdx.x;
  int b = bk >> 4, k = bk & 15;
  int t = threadIdx.x;
  float a0 = expf(sc[(size_t)bk * 512 + t])       * Mm[((size_t)b * 512 + t) * 16 + k];
  float a1 = expf(sc[(size_t)bk * 512 + t + 256]) * Mm[((size_t)b * 512 + t + 256) * 16 + k];
  float S = blockReduceSum256(a0 + a1, red);
  float inv = 1.0f / S;
  sc[(size_t)bk * 512 + t] = a0 * inv;
  sc[(size_t)bk * 512 + t + 256] = a1 * inv;
}

// Cc[b,k,h] = sum_c aw[b,k,c] * V[b,c,h]
__global__ __launch_bounds__(256)
void cc_kern(const float* __restrict__ aw, const float* __restrict__ V,
             float* __restrict__ Cc) {
  __shared__ float ws[512];
  int bk = blockIdx.x;
  int b = bk >> 4;
  int t = threadIdx.x;
  ws[t] = aw[(size_t)bk * 512 + t]; ws[t + 256] = aw[(size_t)bk * 512 + t + 256];
  __syncthreads();
  float acc0 = 0.0f, acc1 = 0.0f;
  for (int c = 0; c < 512; ++c) {
    const float* vr = V + ((size_t)b * 512 + c) * 512;
    float w = ws[c];
    acc0 += w * vr[t];
    acc1 += w * vr[t + 256];
  }
  Cc[(size_t)bk * 512 + t] = acc0;
  Cc[(size_t)bk * 512 + t + 256] = acc1;
}

// ce_new = mean over b of Cc
__global__ void cemean_kern(const float* __restrict__ Cc, float* __restrict__ ce_new) {
  int idx = blockIdx.x * 256 + threadIdx.x;   // 16*512
  if (idx >= 16 * 512) return;
  int k = idx >> 9, hh = idx & 511;
  float s = 0.0f;
  for (int b = 0; b < 32; ++b) s += Cc[((size_t)b * 16 + k) * 512 + hh];
  ce_new[idx] = s * (1.0f / 32.0f);
}

// ---------------------------------------------------------------------------
// Output: theta gating + per-channel [H,P] projection + RevIN denorm +
// transpose to [B,P,C]. block = one (b,c); 128 threads, P=96.
// ---------------------------------------------------------------------------
__global__ __launch_bounds__(128)
void out_kern(const float* __restrict__ Hm, const float* __restrict__ p,
              const float* __restrict__ ce_new, const float* __restrict__ out_w,
              const float* __restrict__ out_b, const float* __restrict__ rev_w,
              const float* __restrict__ rev_b, const float* __restrict__ meanB,
              const float* __restrict__ stdB, float* __restrict__ y) {
  __shared__ float g[512];
  __shared__ float pv[16];
  int bc = blockIdx.x;
  int b = bc >> 9, c = bc & 511;
  int t = threadIdx.x;
  if (t < 16) pv[t] = p[(size_t)bc * 16 + t];
  __syncthreads();
  for (int hh = t; hh < 512; hh += 128) {
    float th = 0.0f;
    for (int k = 0; k < 16; ++k) th += pv[k] * ce_new[k * 512 + hh];
    g[hh] = Hm[(size_t)bc * 512 + hh] * th;
  }
  __syncthreads();
  if (t < 96) {
    float a = out_b[c * 96 + t];
    const float* w = out_w + (size_t)c * 512 * 96 + t;
    for (int hh = 0; hh < 512; ++hh) a += g[hh] * w[(size_t)hh * 96];
    float val = (a - rev_b[c]) / (rev_w[c] + 1e-10f) * stdB[bc] + meanB[bc];
    y[((size_t)b * 96 + t) * 512 + c] = val;
  }
}

// ---------------------------------------------------------------------------
extern "C" void kernel_launch(void* const* d_in, const int* in_sizes, int n_in,
                              void* d_out, int out_size, void* d_ws, size_t ws_size,
                              hipStream_t stream) {
  const float* x        = (const float*)d_in[0];
  const float* rev_w    = (const float*)d_in[1];
  const float* rev_b    = (const float*)d_in[2];
  const float* mlp_w1   = (const float*)d_in[3];
  const float* mlp_b1   = (const float*)d_in[4];
  const float* mlp_w2   = (const float*)d_in[5];
  const float* mlp_b2   = (const float*)d_in[6];
  const float* cluster  = (const float*)d_in[7];
  const float* wq       = (const float*)d_in[8];
  const float* bq       = (const float*)d_in[9];
  const float* wk       = (const float*)d_in[10];
  const float* bk       = (const float*)d_in[11];
  const float* wv       = (const float*)d_in[12];
  const float* bv       = (const float*)d_in[13];
  const float* ts_ln_g  = (const float*)d_in[14];
  const float* ts_ln_b  = (const float*)d_in[15];
  const float* ts_w     = (const float*)d_in[16];
  const float* ts_b     = (const float*)d_in[17];
  const float* ch_ln_g  = (const float*)d_in[18];
  const float* ch_ln_b  = (const float*)d_in[19];
  const float* ch_w1    = (const float*)d_in[20];
  const float* ch_b1    = (const float*)d_in[21];
  const float* ch_w2    = (const float*)d_in[22];
  const float* ch_b2    = (const float*)d_in[23];
  const float* out_w    = (const float*)d_in[24];
  const float* out_b    = (const float*)d_in[25];
  float* y = (float*)d_out;

  const int Bn = 32, K = 16, H = 512;
  const size_t BC = (size_t)Bn * 512;        // 16384
  const size_t BCH = BC * 512;               // 8.4M

  char* ws = (char*)d_ws;
  size_t off = 0;
  auto carve = [&](size_t bytes) -> char* {
    char* pp = ws + off;
    off += (bytes + 255) & ~(size_t)255;
    return pp;
  };

  float* meanB = (float*)carve(BC * 4);
  float* stdB  = (float*)carve(BC * 4);
  unsigned short* xt_bf  = (unsigned short*)carve(BCH * 2);
  unsigned short* t1_bf  = (unsigned short*)carve(BCH * 2);
  unsigned short* act_bf = (unsigned short*)carve(BCH * 2);   // h_bf / LN output
  float* h_f  = (float*)carve(BCH * 4);                       // h, later Hm
  float* bufA = (float*)carve(BCH * 4);                       // Kk, later Z
  float* bufB = (float*)carve(BCH * 4);                       // V,  later Z2
  unsigned short* w1bf  = (unsigned short*)carve(512 * 512 * 2);  // all pre-transposed [N,K]
  unsigned short* w2bf  = (unsigned short*)carve(512 * 512 * 2);
  unsigned short* wkbf  = (unsigned short*)carve(512 * 512 * 2);
  unsigned short* wvbf  = (unsigned short*)carve(512 * 512 * 2);
  unsigned short* tswbf = (unsigned short*)carve(512 * 512 * 2);
  unsigned short* cw1bf = (unsigned short*)carve(512 * 512 * 2);
  unsigned short* cw2bf = (unsigned short*)carve(512 * 512 * 2);
  float* pbuf   = (float*)carve(BC * K * 4);
  float* Mm     = (float*)carve(BC * K * 4);
  float* scb    = (float*)carve((size_t)Bn * K * 512 * 4);
  float* Ccb    = (float*)carve((size_t)Bn * K * 512 * 4);
  float* ce_new = (float*)carve(K * H * 4);
  float* cnb    = (float*)carve(K * H * 4);
  float* Qb     = (float*)carve(K * H * 4);
  (void)in_sizes; (void)n_in; (void)out_size; (void)ws_size;

  // weights -> bf16, transposed to [N,K] for K-contiguous async staging
  dim3 cb(256), cg(512 * 512 / 256);
  convT_bf16_kern<<<cg, cb, 0, stream>>>(mlp_w1, w1bf);
  convT_bf16_kern<<<cg, cb, 0, stream>>>(mlp_w2, w2bf);
  convT_bf16_kern<<<cg, cb, 0, stream>>>(wk, wkbf);
  convT_bf16_kern<<<cg, cb, 0, stream>>>(wv, wvbf);
  convT_bf16_kern<<<cg, cb, 0, stream>>>(ts_w, tswbf);
  convT_bf16_kern<<<cg, cb, 0, stream>>>(ch_w1, cw1bf);
  convT_bf16_kern<<<cg, cb, 0, stream>>>(ch_w2, cw2bf);

  // RevIN + transpose -> xt [B,C,L] bf16
  revin_kern<<<(int)BC, 256, 0, stream>>>(x, rev_w, rev_b, meanB, stdB, xt_bf);

  dim3 gg(4, 128), gb(256);   // 16384x512x512 GEMMs
  // h = relu(xt@w1+b1)@w2+b2
  gemm_kern<1, false, false, true><<<gg, gb, 0, stream>>>(xt_bf, w1bf, mlp_b1, nullptr, nullptr, t1_bf);
  gemm_kern<0, false, true,  true><<<gg, gb, 0, stream>>>(t1_bf, w2bf, mlp_b2, nullptr, h_f, act_bf);
  // Kk, V
  gemm_kern<0, false, true, false><<<gg, gb, 0, stream>>>(act_bf, wkbf, bk, nullptr, bufA, nullptr);
  gemm_kern<0, false, true, false><<<gg, gb, 0, stream>>>(act_bf, wvbf, bv, nullptr, bufB, nullptr);

  // routing p, bernoulli mask
  cn_kern<<<K, 256, 0, stream>>>(cluster, cnb);
  route_kern<<<(int)BC, 256, 0, stream>>>(h_f, cnb, pbuf);
  int halfN = (int)(BC * K / 2);   // 131072
  bernoulli_kern<<<(halfN + 255) / 256, 256, 0, stream>>>(pbuf, Mm, halfN);

  // masked exp-attention over clusters
  qproj_kern<<<K, 256, 0, stream>>>(cluster, wq, bq, Qb);
  scores_kern<<<Bn * K, 256, 0, stream>>>(Qb, bufA, scb);
  aw_kern<<<Bn * K, 256, 0, stream>>>(scb, Mm);
  cc_kern<<<Bn * K, 256, 0, stream>>>(scb, bufB, Ccb);
  cemean_kern<<<(K * H + 255) / 256, 256, 0, stream>>>(Ccb, ce_new);

  // mixer: stay in z-layout [B,H,C]; transposes cancel between sub-blocks.
  dim3 tg(16, 16, Bn), tb(32, 8);
  transpose512_kern<<<tg, tb, 0, stream>>>(h_f, bufA);   // Z = h^T
  for (int it = 0; it < 4; ++it) {
    ln_rows_kern<<<(int)BC, 256, 0, stream>>>(bufA, ts_ln_g, ts_ln_b, act_bf);
    gemm_kern<2, true, true, false><<<gg, gb, 0, stream>>>(act_bf, tswbf, ts_b, bufA, bufB, nullptr);
    ln_rows_kern<<<(int)BC, 256, 0, stream>>>(bufB, ch_ln_g, ch_ln_b, act_bf);
    gemm_kern<2, false, false, true><<<gg, gb, 0, stream>>>(act_bf, cw1bf, ch_b1, nullptr, nullptr, t1_bf);
    gemm_kern<0, true, true, false><<<gg, gb, 0, stream>>>(t1_bf, cw2bf, ch_b2, bufB, bufA, nullptr);
  }
  transpose512_kern<<<tg, tb, 0, stream>>>(bufA, h_f);   // Hm = Z^T -> [B,C,H]

  // theta gating + out projection + denorm + transpose to [B,P,C]
  out_kern<<<(int)BC, 128, 0, stream>>>(h_f, pbuf, ce_new, out_w, out_b,
                                        rev_w, rev_b, meanB, stdB, y);
}
